// RankInfoNCELoss_32538672234794
// MI455X (gfx1250) — compile-verified, matrix operand order checked by reading, not run on hardware
//
#include <hip/hip_runtime.h>

typedef float  v2f  __attribute__((ext_vector_type(2)));
typedef float  v4f  __attribute__((ext_vector_type(4)));
typedef float  v8f  __attribute__((ext_vector_type(8)));
typedef __bf16 v4bf __attribute__((ext_vector_type(4)));
typedef __bf16 v8bf __attribute__((ext_vector_type(8)));
typedef __bf16 v16bf __attribute__((ext_vector_type(16)));

#define BATCH 4096
#define DIM   1024
#define NGRP  1024
#define INV_T 10.0f
#define RANK_W 0.1f
#define SCORE_W 0.01f

// workspace layout (float offsets within the first 64KB)
#define WS_GMAX 0
#define WS_GMIN 1024
#define WS_WSUM 2048
#define WS_GCNT 3072
#define WS_W    4096
#define WS_ROW  8192           // per-row loss [BATCH]
#define WS_HIT  (8192 + 4096)  // per-row argmax-hit flag [BATCH]
#define WS_BF16_BYTES ((size_t)BATCH * DIM * 2)   // one bf16 matrix = 8MB
#define WS_BF16_BASE  ((size_t)65536)             // bf16 region starts after float region

// ---------------- group stats ----------------

__global__ void k_init(float* ws) {
  int i = blockIdx.x * 256 + threadIdx.x;
  if (i < NGRP) {
    ((int*)ws)[WS_GMAX + i] = 0;           // bits of 0.0f; scores >= 0
    ((int*)ws)[WS_GMIN + i] = 0x7f800000;  // +inf
    ws[WS_WSUM + i] = 0.0f;
    ws[WS_GCNT + i] = 0.0f;
  }
}

__global__ void k_stats(const int* __restrict__ qid, const float* __restrict__ scores,
                        float* ws) {
  int i = blockIdx.x * 256 + threadIdx.x;
  if (i < BATCH) {
    int g = qid[i];
    int sb = __float_as_int(scores[i]);  // scores >= 0 -> int order == float order
    atomicMax((int*)ws + WS_GMAX + g, sb);
    atomicMin((int*)ws + WS_GMIN + g, sb);
    atomicAdd(ws + WS_GCNT + g, 1.0f);
  }
}

__global__ void k_weight(const int* __restrict__ qid, const float* __restrict__ ranks,
                         const float* __restrict__ scores, float* ws) {
  int i = blockIdx.x * 256 + threadIdx.x;
  if (i < BATCH) {
    int g = qid[i];
    float gmax = __int_as_float(((const int*)ws)[WS_GMAX + g]);
    float gmin = __int_as_float(((const int*)ws)[WS_GMIN + g]);
    float denom = gmax - gmin;
    float norm = (denom > 0.0f) ? (scores[i] - gmin) / denom : 0.0f;
    float w = __expf(-RANK_W * ranks[i]) * __expf(SCORE_W * norm);
    ws[WS_W + i] = w;
    atomicAdd(ws + WS_WSUM + g, w);
  }
}

__global__ void k_wnorm(const int* __restrict__ qid, float* ws) {
  int i = blockIdx.x * 256 + threadIdx.x;
  if (i < BATCH) {
    int g = qid[i];
    float wsum = ws[WS_WSUM + g];
    ws[WS_W + i] = (wsum > 0.0f) ? ws[WS_W + i] / wsum : 1.0f / ws[WS_GCNT + g];
  }
}

// ---------------- f32 -> bf16 hi/lo split (x = hi + lo, err ~2^-18) ----------------

__global__ void k_split(const float* __restrict__ src, __bf16* __restrict__ dh,
                        __bf16* __restrict__ dl, int n4) {
  int i = blockIdx.x * 256 + threadIdx.x;
  if (i < n4) {
    v4f x = *(const v4f*)(src + (size_t)i * 4);
    v4bf h, l;
#pragma unroll
    for (int c = 0; c < 4; ++c) {
      __bf16 hc = (__bf16)x[c];
      h[c] = hc;
      l[c] = (__bf16)(x[c] - (float)hc);
    }
    *(v4bf*)(dh + (size_t)i * 4) = h;
    *(v4bf*)(dl + (size_t)i * 4) = l;
  }
}

// ---------------- shared epilogue helpers ----------------

__device__ __forceinline__ void fold_tile(const v8f* acc, int cb, int l15,
                                          const float* __restrict__ w,
                                          const int* __restrict__ qid,
                                          const int* qrow, float* m, float* s, float* p,
                                          float* wr, float* bv, int* bi) {
#pragma unroll
  for (int t = 0; t < 8; ++t) {
    int col  = cb + t * 16 + l15;
    float wc = w[col];
    int   qc = qid[col];
#pragma unroll
    for (int j = 0; j < 8; ++j) {
      float v = acc[t][j] * INV_T;
      if (qc == qrow[j]) { p[j] += wc * v; wr[j] += wc; }
      float mn = fmaxf(m[j], v);
      s[j] = s[j] * __expf(m[j] - mn) + __expf(v - mn);
      m[j] = mn;
      if (v > bv[j]) { bv[j] = v; bi[j] = col; }  // strict > keeps first max (cols ascend)
    }
  }
}

__device__ __forceinline__ void epilogue(int rbase, int half, int l15,
                                         const int* __restrict__ qid, float* __restrict__ ws,
                                         const int* qrow, float* m, float* s, float* p,
                                         float* wr, float* bv, int* bi) {
  // reduce across the 16-lane half-group (offsets < 16 stay within halves)
#pragma unroll
  for (int j = 0; j < 8; ++j) {
#pragma unroll
    for (int off = 8; off >= 1; off >>= 1) {
      float m2  = __shfl_xor(m[j],  off, 32);
      float s2  = __shfl_xor(s[j],  off, 32);
      float p2  = __shfl_xor(p[j],  off, 32);
      float w2  = __shfl_xor(wr[j], off, 32);
      float bv2 = __shfl_xor(bv[j], off, 32);
      int   bi2 = __shfl_xor(bi[j], off, 32);
      float mn = fmaxf(m[j], m2);
      s[j] = s[j] * __expf(m[j] - mn) + s2 * __expf(m2 - mn);
      m[j] = mn;
      p[j] += p2; wr[j] += w2;
      if (bv2 > bv[j] || (bv2 == bv[j] && bi2 < bi[j])) { bv[j] = bv2; bi[j] = bi2; }
    }
  }
  if (l15 == 0) {
#pragma unroll
    for (int j = 0; j < 8; ++j) {
      int row = rbase + j + 8 * half;
      ws[WS_ROW + row] = (m[j] + __logf(s[j])) * wr[j] - p[j];
      ws[WS_HIT + row] = (qid[bi[j]] == qrow[j]) ? 1.0f : 0.0f;
    }
  }
}

// ---------------- bf16-split fused GEMM + online softmax + loss ----------------
// block: 128 threads = 4 waves; wave w owns rows [blk*64 + 16w, +16).
// 32 column tiles of 128; K-chunks of 64 (hi+lo) double-buffered via
// global_load_async_to_lds_b128: issue chunk i+1, s_wait_asynccnt 16 retires
// chunk i (async loads complete in order), barrier, compute chunk i.
// LDS col stride 72 bf16 (=36 dwords): frag b128 base banks 36*l15+4h cover
// all 64 banks exactly twice -> conflict-free.

#define LDS_S 72
#define NCHUNK ((BATCH / 128) * (DIM / 64))  // 512 flattened (col-tile, K-chunk) steps

__device__ __forceinline__ v16bf frag16(const __bf16* base, int half) {
  // A/B 16-bit fragment: elems 0..7 = K (8*half)+0..7, elems 8..15 = K 16+(8*half)+0..7
  v8bf a0 = *(const v8bf*)(base + 8 * half);
  v8bf a1 = *(const v8bf*)(base + 16 + 8 * half);
  return __builtin_shufflevector(a0, a1, 0, 1, 2, 3, 4, 5, 6, 7,
                                 8, 9, 10, 11, 12, 13, 14, 15);
}

__device__ __forceinline__ void async_ld16(const __bf16* g, const __bf16* lds) {
  unsigned int  loff = (unsigned int)(unsigned long long)lds;  // low 32b of flat = LDS offset
  unsigned long long ga = (unsigned long long)g;
  asm volatile("global_load_async_to_lds_b128 %0, %1, off"
               :: "v"(loff), "v"(ga) : "memory");
}

__launch_bounds__(128)
__global__ void k_main_bf16(const __bf16* __restrict__ qh, const __bf16* __restrict__ ql,
                            const __bf16* __restrict__ ah, const __bf16* __restrict__ al,
                            const int* __restrict__ qid, float* __restrict__ ws) {
  __shared__ __bf16 lds_h[2][128 * LDS_S];
  __shared__ __bf16 lds_l[2][128 * LDS_S];

  const float* w = ws + WS_W;
  const int tid  = threadIdx.x;
  const int lane = tid & 31;
  const int wave = tid >> 5;
  const int half = lane >> 4;
  const int l15  = lane & 31 & 15;
  const int rbase = blockIdx.x * 64 + wave * 16;

  int   qrow[8];
  float m[8], s[8], p[8], wr[8], bv[8];
  int   bi[8];
#pragma unroll
  for (int j = 0; j < 8; ++j) {
    qrow[j] = qid[rbase + j + 8 * half];
    m[j] = -3.4e38f; s[j] = 0.0f; p[j] = 0.0f; wr[j] = 0.0f;
    bv[j] = -3.4e38f; bi[j] = 0;
  }

  const __bf16* qrh = qh + (size_t)(rbase + l15) * DIM;
  const __bf16* qrl = ql + (size_t)(rbase + l15) * DIM;

  // per-thread staging geometry: 8 b128 slots per matrix per chunk
  const int scol = tid >> 3;          // col = scol + 16*i
  const int sk8  = (tid & 7) << 3;    // 0..56

  // prologue: issue chunk 0 (cb=0, kb=0) into buffer 0
#pragma unroll
  for (int i = 0; i < 8; ++i) {
    int col = scol + 16 * i;
    size_t gsrc = (size_t)col * DIM + sk8;
    int    ldst = col * LDS_S + sk8;
    async_ld16(ah + gsrc, &lds_h[0][ldst]);
    async_ld16(al + gsrc, &lds_l[0][ldst]);
  }

  for (int cb_i = 0; cb_i < BATCH / 128; ++cb_i) {
    const int cb = cb_i * 128;
    v8f acc[8];
    const v8f vz = {};
#pragma unroll
    for (int t = 0; t < 8; ++t) acc[t] = vz;

    for (int kb_i = 0; kb_i < DIM / 64; ++kb_i) {
      const int kb  = kb_i * 64;
      const int c   = cb_i * (DIM / 64) + kb_i;
      const int par = kb_i & 1;

      if (c + 1 < NCHUNK) {  // issue next chunk into the other buffer
        const int nc  = c + 1;
        const int ncb = (nc >> 4) * 128;
        const int nkb = (nc & 15) * 64;
        const int np  = nc & 1;
#pragma unroll
        for (int i = 0; i < 8; ++i) {
          int col = scol + 16 * i;
          size_t gsrc = (size_t)(ncb + col) * DIM + nkb + sk8;
          int    ldst = col * LDS_S + sk8;
          async_ld16(ah + gsrc, &lds_h[np][ldst]);
          async_ld16(al + gsrc, &lds_l[np][ldst]);
        }
        // 16 newest ops may remain outstanding -> previous chunk's 16 retired
        asm volatile("s_wait_asynccnt 0x10" ::: "memory");
      } else {
        asm volatile("s_wait_asynccnt 0x0" ::: "memory");
      }
      __syncthreads();  // all waves' writes for chunk c are visible

#pragma unroll
      for (int ks = 0; ks < 64; ks += 32) {
        v16bf afh = frag16(qrh + kb + ks, half);
        v16bf afl = frag16(qrl + kb + ks, half);
#pragma unroll
        for (int t = 0; t < 8; ++t) {
          v16bf bfh = frag16(&lds_h[par][(t * 16 + l15) * LDS_S + ks], half);
          v16bf bfl = frag16(&lds_l[par][(t * 16 + l15) * LDS_S + ks], half);
          acc[t] = __builtin_amdgcn_wmma_f32_16x16x32_bf16(
              false, afh, false, bfh, (short)0, acc[t], false, false);
          acc[t] = __builtin_amdgcn_wmma_f32_16x16x32_bf16(
              false, afh, false, bfl, (short)0, acc[t], false, false);
          acc[t] = __builtin_amdgcn_wmma_f32_16x16x32_bf16(
              false, afl, false, bfh, (short)0, acc[t], false, false);
        }
      }
      __syncthreads();  // everyone done reading buf[par] before it is re-issued
    }
    fold_tile(acc, cb, l15, w, qid, qrow, m, s, p, wr, bv, bi);
  }
  epilogue(rbase, half, l15, qid, ws, qrow, m, s, p, wr, bv, bi);
}

// ---------------- f32 WMMA fallback (used if workspace too small) ----------------

#define LDS_STRIDE 68

__launch_bounds__(128)
__global__ void k_main_f32(const float* __restrict__ q, const float* __restrict__ a,
                           const int* __restrict__ qid, float* __restrict__ ws) {
  __shared__ float lds_a[128 * LDS_STRIDE];

  const float* w = ws + WS_W;
  const int tid  = threadIdx.x;
  const int lane = tid & 31;
  const int wave = tid >> 5;
  const int half = lane >> 4;
  const int l15  = lane & 15;
  const int rbase = blockIdx.x * 64 + wave * 16;

  int   qrow[8];
  float m[8], s[8], p[8], wr[8], bv[8];
  int   bi[8];
#pragma unroll
  for (int j = 0; j < 8; ++j) {
    qrow[j] = qid[rbase + j + 8 * half];
    m[j] = -3.4e38f; s[j] = 0.0f; p[j] = 0.0f; wr[j] = 0.0f;
    bv[j] = -3.4e38f; bi[j] = 0;
  }
  const float* qrowptr = q + (size_t)(rbase + l15) * DIM + 2 * half;

  for (int cb = 0; cb < BATCH; cb += 128) {
    v8f acc[8];
    const v8f vz = {};
#pragma unroll
    for (int t = 0; t < 8; ++t) acc[t] = vz;

    for (int kb = 0; kb < DIM; kb += 64) {
      __syncthreads();
#pragma unroll
      for (int i = 0; i < 16; ++i) {
        int f   = tid + 128 * i;
        int col = f >> 4;
        int k4  = (f & 15) << 2;
        *(v4f*)(&lds_a[col * LDS_STRIDE + k4]) =
            *(const v4f*)(a + (size_t)(cb + col) * DIM + kb + k4);
      }
      __syncthreads();
#pragma unroll
      for (int kk = 0; kk < 64; kk += 4) {
        v2f af = *(const v2f*)(qrowptr + kb + kk);
#pragma unroll
        for (int t = 0; t < 8; ++t) {
          v2f bf = *(const v2f*)(&lds_a[(t * 16 + l15) * LDS_STRIDE + kk + 2 * half]);
          acc[t] = __builtin_amdgcn_wmma_f32_16x16x4_f32(
              false, af, false, bf, (short)0, acc[t], false, false);
        }
      }
    }
    fold_tile(acc, cb, l15, w, qid, qrow, m, s, p, wr, bv, bi);
  }
  epilogue(rbase, half, l15, qid, ws, qrow, m, s, p, wr, bv, bi);
}

// ---------------- deterministic final reduction ----------------

__global__ void k_final(const float* __restrict__ ws, float* __restrict__ out) {
  __shared__ float sl[256], sa[256];
  int t = threadIdx.x;
  float l = 0.0f, a = 0.0f;
  for (int i = 0; i < 16; ++i) {
    l += ws[WS_ROW + t * 16 + i];
    a += ws[WS_HIT + t * 16 + i];
  }
  sl[t] = l; sa[t] = a;
  __syncthreads();
  for (int o = 128; o > 0; o >>= 1) {
    if (t < o) { sl[t] += sl[t + o]; sa[t] += sa[t + o]; }
    __syncthreads();
  }
  if (t == 0) {
    out[0] = sl[0] / (float)BATCH;
    out[1] = sa[0] / (float)BATCH;
  }
}

extern "C" void kernel_launch(void* const* d_in, const int* in_sizes, int n_in,
                              void* d_out, int out_size, void* d_ws, size_t ws_size,
                              hipStream_t stream) {
  const float* qe     = (const float*)d_in[0];
  const float* ae     = (const float*)d_in[1];
  const int*   qid    = (const int*)d_in[2];
  const float* ranks  = (const float*)d_in[3];
  const float* scores = (const float*)d_in[4];
  float* out = (float*)d_out;
  float* ws  = (float*)d_ws;

  k_init  <<<NGRP / 256, 256, 0, stream>>>(ws);
  k_stats <<<BATCH / 256, 256, 0, stream>>>(qid, scores, ws);
  k_weight<<<BATCH / 256, 256, 0, stream>>>(qid, ranks, scores, ws);
  k_wnorm <<<BATCH / 256, 256, 0, stream>>>(qid, ws);

  const size_t need = WS_BF16_BASE + 4 * WS_BF16_BYTES;
  if (ws_size >= need) {
    char* base = (char*)d_ws + WS_BF16_BASE;
    __bf16* qh = (__bf16*)(base);
    __bf16* ql = (__bf16*)(base + WS_BF16_BYTES);
    __bf16* ah = (__bf16*)(base + 2 * WS_BF16_BYTES);
    __bf16* al = (__bf16*)(base + 3 * WS_BF16_BYTES);
    const int n4 = BATCH * DIM / 4;
    k_split<<<n4 / 256, 256, 0, stream>>>(qe, qh, ql, n4);
    k_split<<<n4 / 256, 256, 0, stream>>>(ae, ah, al, n4);
    k_main_bf16<<<BATCH / 64, 128, 0, stream>>>(qh, ql, ah, al, qid, ws);
  } else {
    k_main_f32<<<BATCH / 64, 128, 0, stream>>>(qe, ae, qid, ws);
  }
  k_final<<<1, 256, 0, stream>>>(ws, out);
}
